// JWHT_42322607734922
// MI455X (gfx1250) — matrix operand order repaired and checked by reading, not compiled, not run on hardware
//
#include <hip/hip_runtime.h>
#include <hip/hip_bf16.h>

typedef _Float16 half_t;
typedef __attribute__((ext_vector_type(16))) _Float16 v16h;
typedef __attribute__((ext_vector_type(8)))  float    v8f;
typedef unsigned int v4u __attribute__((ext_vector_type(4)));
typedef int          v8i __attribute__((ext_vector_type(8)));
typedef int          v4i __attribute__((ext_vector_type(4)));

#define NTHREADS 256
#define TM       128     // rows per block
#define LDH0     520     // f16 ld, 512-wide stages (stride 260 dwords == 4 mod 64 -> conflict-free)
#define LDH2     136
#define LDG3     68
#define LDH3     72
#define LDH4     136
#define LDG5     36
#define LDH5     40
#define R0_BYTES   (TM * LDH0 * 2)            // 133120
#define BS_BYTES   (16 * (512 + 8) * 2)       // 16640 (max panel: K=512)
#define BS0_OFF    (2 * R0_BYTES)             // 266240
#define BS1_OFF    (BS0_OFF + BS_BYTES)       // 282880
#define SMEM_BYTES (BS1_OFF + BS_BYTES)       // 299520 < 320KB/WGP

// ---- Tensor Data Mover availability probe ----
#if defined(__has_builtin)
#  if __has_builtin(__builtin_amdgcn_tensor_load_to_lds) && __has_builtin(__builtin_amdgcn_s_wait_tensorcnt)
#    define USE_TDM 1
#  endif
#endif
#ifndef USE_TDM
#  define USE_TDM 0
#endif

union FragU { v16h v; uint4 q[2]; };

// ---- A fragment: 16x32 f16, rows m0.., cols k0.., row-major LDS (ISA 16-bit A layout) ----
// Per lane: two contiguous 16B runs -> 2x ds_load_b128.
//   lanes 0-15 : halves [k0 .. k0+7]   and [k0+16 .. k0+23]
//   lanes 16-31: halves [k0+8 .. k0+15] and [k0+24 .. k0+31]
__device__ __forceinline__ v16h load_frag_A(const half_t* A, int lda, int m0, int k0, int lane) {
    const int hb = lane >> 4, mr = lane & 15;
    const half_t* p = A + (m0 + mr) * lda + k0 + hb * 8;
    FragU f;
    f.q[0] = *(const uint4*)(p);
    f.q[1] = *(const uint4*)(p + 16);
    return f.v;
}

// ---- B fragment: 32x16 f16 from LDS-staged panel Bs[16][ldb] (row n = output col) ----
// lanes 0-15: K=k0..k0+7 and k0+16..k0+23; lanes 16-31: K=k0+8..k0+15 and k0+24..k0+31; 2x ds_load_b128.
__device__ __forceinline__ v16h load_frag_B_lds(const half_t* Bs, int ldb, int k0, int lane) {
    const int hb = lane >> 4, nc = lane & 15;
    const half_t* p = Bs + nc * ldb + k0 + hb * 8;
    FragU f;
    f.q[0] = *(const uint4*)(p);
    f.q[1] = *(const uint4*)(p + 16);
    return f.v;
}

// ---- TDM: stage one 16-row x K-half panel of Wt[N][K] into LDS at byte offset lds_off,
//      padding each row (K halves = 2^(padv+1) dwords) by 8 halves (4 dwords) ----
template<int K>
__device__ __forceinline__ void tdm_load_chunk(const half_t* Wt, int chunk, unsigned lds_off) {
#if USE_TDM
    constexpr unsigned padv = (K == 512) ? 7u : (K == 128) ? 5u : (K == 64) ? 4u : 3u;
    unsigned long long ga = (unsigned long long)(size_t)(const void*)(Wt + (size_t)chunk * 16 * K);
    v4u g0;
    g0[0] = 1u;                                   // count=1, user descriptor, no gather
    g0[1] = lds_off;                              // lds_addr (dynamic LDS starts at offset 0)
    g0[2] = (unsigned)ga;                         // global_addr[31:0]
    g0[3] = (unsigned)((ga >> 32) & 0x1ffffffull) | (2u << 30);   // addr[56:32] | type=2
    v8i g1;
    g1[0] = (int)((1u << 16) | (1u << 20) | (padv << 22) | (3u << 25)); // data_size=2B, pad_en, pad cfg
    g1[1] = (int)((unsigned)(K & 0xffff) << 16);  // tensor_dim0[15:0] in bits 63:48
    g1[2] = (int)(16u << 16);                     // tensor_dim1 = 16 rows (== tile, no OOB)
    g1[3] = (int)((unsigned)K << 16);             // tile_dim0 = K
    g1[4] = 16;                                   // tile_dim1 = 16
    g1[5] = K;                                    // tensor_dim0_stride = K
    g1[6] = 0;
    g1[7] = 0;
    v4i gz = (v4i){0, 0, 0, 0};
#if __clang_major__ >= 23
    v8i gz8 = (v8i){0, 0, 0, 0, 0, 0, 0, 0};
    __builtin_amdgcn_tensor_load_to_lds(g0, g1, gz, gz, gz8, 0);
#else
    __builtin_amdgcn_tensor_load_to_lds(g0, g1, gz, gz, 0);
#endif
#endif
}

// ---- fallback cooperative synchronous panel copy (used if TDM builtin absent) ----
template<int K>
__device__ __forceinline__ void copy_chunk_sync(const half_t* Wt, int chunk, half_t* Bs, int tid) {
    constexpr int LDB = K + 8;
    const unsigned int* src = (const unsigned int*)(Wt + (size_t)chunk * 16 * K);
    unsigned int* dst = (unsigned int*)Bs;
    constexpr int DW = 8 * K;                     // 16*K/2 dwords
    for (int j = tid; j < DW; j += NTHREADS) {
        int n = j / (K / 2), d = j % (K / 2);
        dst[n * (LDB / 2) + d] = src[n * (K / 2) + d];
    }
}

// ---- staged GEMM: out = act(A[TM x K] @ Wt^T + bias); B panels double-buffered via TDM ----
// A fragments are hoisted into registers once (immune to the per-chunk barriers), so the
// steady-state LDS traffic is the B fragments only.
// MODE 0: f16 LDS + relu; MODE 1: f32 LDS; MODE 2: f32 global
template<int K, int N, int MODE>
__device__ __forceinline__ void staged_gemm(const half_t* __restrict__ Wt,
                                            const half_t* __restrict__ A, int lda,
                                            char* smem,
                                            half_t* Of16, int ldo16,
                                            float* Of32, int ldo32,
                                            float* __restrict__ gout, int rbase,
                                            const float* __restrict__ bias,
                                            int wid, int lane, int tid) {
    constexpr int LDB = K + 8;
    constexpr int NC  = N / 16;
    constexpr int KT  = K / 32;
    const int m0 = wid * 16;
    const int hb = lane >> 4, nc = lane & 15;

    // hoist the wave's full A strip into registers (K=512 -> 128 VGPRs)
    v16h afr[KT];
#pragma unroll
    for (int kt = 0; kt < KT; ++kt)
        afr[kt] = load_frag_A(A, lda, m0, kt * 32, lane);

#if USE_TDM
    if (wid == 0) {
        tdm_load_chunk<K>(Wt, 0, BS0_OFF);
        __builtin_amdgcn_s_wait_tensorcnt(0);
    }
    __syncthreads();
#endif
    for (int c = 0; c < NC; ++c) {
#if USE_TDM
        const half_t* Bs = (const half_t*)(smem + ((c & 1) ? BS1_OFF : BS0_OFF));
        if (wid == 0 && c + 1 < NC)
            tdm_load_chunk<K>(Wt, c + 1, (c & 1) ? BS0_OFF : BS1_OFF);
#else
        half_t* BsW = (half_t*)(smem + BS0_OFF);
        copy_chunk_sync<K>(Wt, c, BsW, tid);
        __syncthreads();
        const half_t* Bs = BsW;
#endif
        v8f acc = {};
#pragma unroll
        for (int kt = 0; kt < KT; ++kt) {
            v16h b = load_frag_B_lds(Bs, LDB, kt * 32, lane);
            acc = __builtin_amdgcn_wmma_f32_16x16x32_f16(false, afr[kt], false, b,
                                                         (short)0, acc, false, false);
        }
        const int n0 = c * 16;
        const float bb = bias[n0 + nc];
        if constexpr (MODE == 0) {
#pragma unroll
            for (int r = 0; r < 8; ++r)
                Of16[(m0 + r + 8 * hb) * ldo16 + n0 + nc] = (half_t)fmaxf(acc[r] + bb, 0.0f);
        } else if constexpr (MODE == 1) {
#pragma unroll
            for (int r = 0; r < 8; ++r)
                Of32[(m0 + r + 8 * hb) * ldo32 + n0 + nc] = acc[r] + bb;
        } else {
#pragma unroll
            for (int r = 0; r < 8; ++r)
                gout[(size_t)(rbase + m0 + r + 8 * hb) * 32 + n0 + nc] = acc[r] + bb;
        }
#if USE_TDM
        if (wid == 0 && c + 1 < NC)
            __builtin_amdgcn_s_wait_tensorcnt(0);
#endif
        __syncthreads();
    }
}

// ---- in-place FWHT butterflies over rows of G[TM][ldg], width n ----
__device__ __forceinline__ void fwht_lds(float* G, int ldg, int n, int tid) {
    const int halfn = n >> 1;
    for (int h = 1; h < n; h <<= 1) {
        __syncthreads();
        for (int idx = tid; idx < TM * halfn; idx += NTHREADS) {
            int row = idx / halfn, j = idx % halfn;
            int p = (j / h) * (2 * h) + (j % h);
            int q = p + h;
            float a = G[row * ldg + p], b = G[row * ldg + q];
            G[row * ldg + p] = a + b;
            G[row * ldg + q] = a - b;
        }
    }
    __syncthreads();
}

__device__ __forceinline__ void conv_f32_to_f16(const float* G, int ldg, half_t* H, int ldh,
                                                int ncols, float scale, float add, int tid) {
    for (int idx = tid; idx < TM * ncols; idx += NTHREADS) {
        int row = idx / ncols, c = idx % ncols;
        H[row * ldh + c] = (half_t)(G[row * ldg + c] * scale + add);
    }
}

// ---- weight prep: fp32 W[K][N] -> f16 Wt[N][K] ----
__global__ void JWHT_prep_kernel(const float* __restrict__ W, half_t* __restrict__ Wt,
                                 int K, int N) {
    int idx = blockIdx.x * blockDim.x + threadIdx.x;
    if (idx < K * N) {
        int k = idx / N, n = idx % N;
        Wt[n * K + k] = (half_t)W[idx];
    }
}

// ---- fused pipeline ----
__global__ __launch_bounds__(NTHREADS)
void JWHT_fused_kernel(const float* __restrict__ x,
                       const float* __restrict__ dw1, const float* __restrict__ dw2,
                       const half_t* __restrict__ eW1t, const float* __restrict__ eb1,
                       const half_t* __restrict__ eW2t, const float* __restrict__ eb2,
                       const half_t* __restrict__ eW3t, const float* __restrict__ eb3,
                       const float* __restrict__ w1c, const float* __restrict__ w1b,
                       const half_t* __restrict__ dW1t, const float* __restrict__ db1,
                       const half_t* __restrict__ dW2t, const float* __restrict__ db2,
                       const float* __restrict__ w2c, const float* __restrict__ w2b,
                       const half_t* __restrict__ pWt, const float* __restrict__ pb,
                       float* __restrict__ out) {
    extern __shared__ char smem[];
    half_t* R0h = (half_t*)smem;
    half_t* R1h = (half_t*)(smem + R0_BYTES);
    float*  R0f = (float*)smem;
    float*  R1f = (float*)(smem + R0_BYTES);

    const int tid  = threadIdx.x;
    const int lane = tid & 31;
    const int wid  = tid >> 5;
    const int rbase = blockIdx.x * TM;

    // ---- stage 0: downsample x (TM x 1024 fp32) -> h0 (TM x 512 f16) in R0 ----
    {
        const float d10 = dw1[0], d11 = dw1[1], d12 = dw1[2], d13 = dw1[3];
        const float d20 = dw2[0], d21 = dw2[1], d22 = dw2[2], d23 = dw2[3];
        const float4* x4 = (const float4*)x;
        for (int i = 0; i < (TM * 256) / NTHREADS; ++i) {
            int g = tid + i * NTHREADS;
            int row = g >> 8, t = g & 255;
            float4 v = x4[(size_t)(rbase + row) * 256 + t];
            float r = v.x * d10 + v.y * d11 + v.z * d12 + v.w * d13;
            float z = v.x * d20 + v.y * d21 + v.z * d22 + v.w * d23;
            R0h[row * LDH0 + t]       = (half_t)r;
            R0h[row * LDH0 + 256 + t] = (half_t)z;
        }
    }
    __syncthreads();

    // ---- GEMM1: h1 = relu(h0 @ eW1 + eb1)   512 -> 512, R0 -> R1 ----
    staged_gemm<512, 512, 0>(eW1t, R0h, LDH0, smem, R1h, LDH0, nullptr, 0,
                             nullptr, 0, eb1, wid, lane, tid);

    // ---- GEMM2: h2 = relu(h1 @ eW2 + eb2)   512 -> 128, R1 -> R0 ----
    staged_gemm<512, 128, 0>(eW2t, R1h, LDH0, smem, R0h, LDH2, nullptr, 0,
                             nullptr, 0, eb2, wid, lane, tid);

    // ---- GEMM3: g3 = h2 @ eW3 + eb3         128 -> 64 (f32), R0 -> R1 ----
    staged_gemm<128, 64, 1>(eW3t, R0h, LDH2, smem, nullptr, 0, R1f, LDG3,
                            nullptr, 0, eb3, wid, lane, tid);

    // ---- FWHT-64, * (w1c/sqrt(64)) + w1b -> h3 f16 in R0 ----
    fwht_lds(R1f, LDG3, 64, tid);
    conv_f32_to_f16(R1f, LDG3, R0h, LDH3, 64, w1c[0] * 0.125f, w1b[0], tid);
    __syncthreads();

    // ---- GEMM4: h4 = relu(h3 @ dW1 + db1)   64 -> 128, R0 -> R1 ----
    staged_gemm<64, 128, 0>(dW1t, R0h, LDH3, smem, R1h, LDH4, nullptr, 0,
                            nullptr, 0, db1, wid, lane, tid);

    // ---- GEMM5: g5 = h4 @ dW2 + db2         128 -> 32 (f32), R1 -> R0 ----
    staged_gemm<128, 32, 1>(dW2t, R1h, LDH4, smem, nullptr, 0, R0f, LDG5,
                            nullptr, 0, db2, wid, lane, tid);

    // ---- FWHT-32, * (w2c/sqrt(32)) + w2b -> h5 f16 in R1 ----
    fwht_lds(R0f, LDG5, 32, tid);
    conv_f32_to_f16(R0f, LDG5, R1h, LDH5, 32, w2c[0] * 0.17677669529663687f, w2b[0], tid);
    __syncthreads();

    // ---- GEMM6: out = h5 @ pW + pb          32 -> 32, f32 to global ----
    staged_gemm<32, 32, 2>(pWt, R1h, LDH5, smem, nullptr, 0, nullptr, 0,
                           out, rbase, pb, wid, lane, tid);
}

extern "C" void kernel_launch(void* const* d_in, const int* in_sizes, int n_in,
                              void* d_out, int out_size, void* d_ws, size_t ws_size,
                              hipStream_t stream) {
    const float* x    = (const float*)d_in[0];
    const float* dw1  = (const float*)d_in[1];
    const float* dw2  = (const float*)d_in[2];
    const float* eW1  = (const float*)d_in[3];
    const float* eb1  = (const float*)d_in[4];
    const float* eW2  = (const float*)d_in[5];
    const float* eb2  = (const float*)d_in[6];
    const float* eW3  = (const float*)d_in[7];
    const float* eb3  = (const float*)d_in[8];
    const float* w1c  = (const float*)d_in[9];
    const float* w1b  = (const float*)d_in[10];
    const float* dW1  = (const float*)d_in[11];
    const float* db1  = (const float*)d_in[12];
    const float* dW2  = (const float*)d_in[13];
    const float* db2  = (const float*)d_in[14];
    const float* w2c  = (const float*)d_in[15];
    const float* w2b  = (const float*)d_in[16];
    const float* pW   = (const float*)d_in[17];
    const float* pb   = (const float*)d_in[18];
    float* out = (float*)d_out;

    char* ws = (char*)d_ws;
    half_t* eW1t = (half_t*)(ws + 0);
    half_t* eW2t = (half_t*)(ws + 524288);
    half_t* eW3t = (half_t*)(ws + 655360);
    half_t* dW1t = (half_t*)(ws + 671744);
    half_t* dW2t = (half_t*)(ws + 688128);
    half_t* pWt  = (half_t*)(ws + 696320);

    JWHT_prep_kernel<<<(512 * 512 + 255) / 256, 256, 0, stream>>>(eW1, eW1t, 512, 512);
    JWHT_prep_kernel<<<(512 * 128 + 255) / 256, 256, 0, stream>>>(eW2, eW2t, 512, 128);
    JWHT_prep_kernel<<<(128 * 64  + 255) / 256, 256, 0, stream>>>(eW3, eW3t, 128, 64);
    JWHT_prep_kernel<<<(64  * 128 + 255) / 256, 256, 0, stream>>>(dW1, dW1t, 64, 128);
    JWHT_prep_kernel<<<(128 * 32  + 255) / 256, 256, 0, stream>>>(dW2, dW2t, 128, 32);
    JWHT_prep_kernel<<<(32  * 32  + 255) / 256, 256, 0, stream>>>(pW,  pWt,  32, 32);

    hipFuncSetAttribute((const void*)JWHT_fused_kernel,
                        hipFuncAttributeMaxDynamicSharedMemorySize, SMEM_BYTES);

    const int nblocks = 65536 / TM;  // 512
    JWHT_fused_kernel<<<nblocks, NTHREADS, SMEM_BYTES, stream>>>(
        x, dw1, dw2, eW1t, eb1, eW2t, eb2, eW3t, eb3, w1c, w1b,
        dW1t, db1, dW2t, db2, w2c, w2b, pWt, pb, out);
}
